// DifferentiableAstar_13271448945030
// MI455X (gfx1250) — matrix operand order repaired and checked by reading, not compiled, not run on hardware
//
#include <hip/hip_runtime.h>
#include <math.h>

// ---------------------------------------------------------------------------
// Differentiable A* (forward values) for B=32, H=W=32, T=256.
// Design: one wave32 per batch; lane == grid row; all mutable state in LDS;
// open/hist/obstacle as per-lane 32-bit masks; selection scores kept
// incrementally so each step = row-scan (ds_load_b128) + shfl butterfly
// argmax (first-index tie-break, matching jnp.argmax) + <=9-cell relax.
// The reference's goal-coordinate einsum is computed with
// v_wmma_f32_16x16x32_f16 into workspace.
// ---------------------------------------------------------------------------

#define HW 1024
#define GRIDW 32
#define T_STEPS 256
// np.float32(np.sqrt(32))
#define SIZE_NORM 5.65685424949238f

typedef __attribute__((ext_vector_type(16))) _Float16 v16h_t;
typedef __attribute__((ext_vector_type(8)))  float    v8f_t;

// ------------------------- WMMA: goal-coordinate einsum ---------------------
// goal_loc[b,k] = sum_cell loc[k][cell] * goal_maps[b][cell]
// A = goal_maps [32 x 1024] (two 16-row M tiles), B = loc^T [1024 x 16]
// (col 0 = row coord i, col 1 = col coord j, cols 2..15 = 0).
// Accumulate K=1024 in 32-wide chunks: 64 v_wmma_f32_16x16x32_f16 total.
__global__ __launch_bounds__(32) void goal_einsum_wmma(
    const float* __restrict__ goal_maps, float* __restrict__ out_coords)
{
  const int lane = (int)threadIdx.x;          // wave32
  const int m    = lane & 15;
  const int nn   = lane & 15;
  for (int tile = 0; tile < 2; ++tile) {
    const int b = tile * 16 + m;
    v8f_t acc = {};
    for (int kc = 0; kc < HW; kc += 32) {
      // A fragment (16x32 f16, ISA 7.12.2): lanes 0-15 -> K kc+0..7 / kc+16..23
      //                                     lanes16-31 -> K kc+8..15 / kc+24..31
      v16h_t a;
      const int kbase = kc + ((lane < 16) ? 0 : 8);
      #pragma unroll
      for (int i = 0; i < 8; ++i) {
        a[i]     = (_Float16)goal_maps[b * HW + kbase + i];
        a[8 + i] = (_Float16)goal_maps[b * HW + kbase + 16 + i];
      }
      // B fragment (32x16 f16): lane n=lane&15; lanes 0-15 hold K kc+0..7 / kc+16..23,
      // lanes 16-31 hold K kc+8..15 / kc+24..31 (same K order as A).
      v16h_t bm;
      const int kb = kc + ((lane < 16) ? 0 : 8);
      #pragma unroll
      for (int i = 0; i < 16; ++i) {
        const int cell = kb + i + ((i < 8) ? 0 : 8);
        const float val = (nn == 0) ? (float)(cell >> 5)
                        : (nn == 1) ? (float)(cell & 31) : 0.0f;
        bm[i] = (_Float16)val;
      }
      acc = __builtin_amdgcn_wmma_f32_16x16x32_f16(
          false, a, false, bm, (short)0, acc, false, false);
    }
    // D 16x16 f32 layout: VGPR r, lane l -> M = r + (l<16?0:8), N = l&15
    if (nn < 2) {
      #pragma unroll
      for (int r = 0; r < 8; ++r) {
        const int mm = r + ((lane < 16) ? 0 : 8);
        out_coords[(tile * 16 + mm) * 2 + nn] = acc[r];
      }
    }
  }
}

// ------------------------------- A* kernel ---------------------------------
__global__ __launch_bounds__(32) void astar_wave_kernel(
    const float* __restrict__ cost_maps,
    const float* __restrict__ start_maps,
    const float* __restrict__ goal_maps,
    const float* __restrict__ obstacles_maps,
    float* __restrict__ out, int B)
{
  __shared__ alignas(16) float s_v[HW];     // f_exp cache (later reused: path)
  __shared__ float s_g[HW];                 // g values
  __shared__ float s_ht[HW];                // heuristic + cost ("h" in ref)
  __shared__ float s_cost[HW];              // cost map
  __shared__ int   s_par[HW];               // parent pointers

  const int b    = (int)blockIdx.x;
  const int lane = (int)threadIdx.x;        // lane == row (W == 32 == wave32)
  const float* costp  = cost_maps      + b * HW;
  const float* startp = start_maps     + b * HW;
  const float* goalp  = goal_maps      + b * HW;
  const float* obstp  = obstacles_maps + b * HW;

  // ---- init: goal index + per-row bitmasks -------------------------------
  int gidx = 1 << 30;
  unsigned obs_mask = 0u, open_mask = 0u, hist_mask = 0u;
  for (int j = 0; j < GRIDW; ++j) {
    const int n = (lane << 5) | j;
    if (goalp[n]  > 0.5f) gidx = n;
    if (obstp[n]  > 0.5f) obs_mask  |= (1u << j);
    if (startp[n] > 0.5f) open_mask |= (1u << j);
  }
  for (int off = 16; off > 0; off >>= 1) {
    const int o = __shfl_xor(gidx, off, 32);
    gidx = (o < gidx) ? o : gidx;
  }
  const float gi_f = (float)(gidx >> 5);
  const float gj_f = (float)(gidx & 31);

  // heuristic + cost; g=0; parents=goal; score cache for open cells
  for (int j = 0; j < GRIDW; ++j) {
    const int n = (lane << 5) | j;
    const float cn = costp[n];
    const float dx = fabsf((float)lane - gi_f);
    const float dy = fabsf((float)j - gj_f);
    const float hmd = __fsub_rn(__fadd_rn(dx, dy), fminf(dx, dy));
    const float euc = __fsqrt_rn(__fadd_rn(__fmul_rn(dx, dx), __fmul_rn(dy, dy)));
    const float ht  = __fadd_rn(__fadd_rn(hmd, __fmul_rn(0.001f, euc)), cn);
    s_cost[n] = cn;
    s_ht[n]   = ht;
    s_g[n]    = 0.0f;
    s_par[n]  = gidx;
    const float f0 = __fmul_rn(0.5f, ht);   // 0.5*g + 0.5*h with g == 0
    s_v[n] = ((open_mask >> j) & 1u) ? expf(__fdiv_rn(-f0, SIZE_NORM)) : 0.0f;
  }
  __syncthreads();

  // ---- 256 sequential A* steps ------------------------------------------
  for (int t = 0; t < T_STEPS; ++t) {
    // argmax over own row (strict > keeps lowest index on ties)
    const float4* v4 = (const float4*)(s_v + (lane << 5));
    float bv = -1.0f;
    int   bi = lane << 5;
    #pragma unroll
    for (int q = 0; q < 8; ++q) {
      const float4 x = v4[q];
      const int nb = (lane << 5) + (q << 2);
      if (x.x > bv) { bv = x.x; bi = nb;     }
      if (x.y > bv) { bv = x.y; bi = nb + 1; }
      if (x.z > bv) { bv = x.z; bi = nb + 2; }
      if (x.w > bv) { bv = x.w; bi = nb + 3; }
    }
    // wave-wide lexicographic max (val desc, index asc) — matches jnp.argmax
    for (int off = 16; off > 0; off >>= 1) {
      const float ov = __shfl_xor(bv, off, 32);
      const int   oi = __shfl_xor(bi, off, 32);
      if (ov > bv || (ov == bv && oi < bi)) { bv = ov; bi = oi; }
    }
    const int s  = bi;
    const int si = s >> 5;
    const int sj = s & 31;
    const float g2v = __fadd_rn(s_g[s], s_cost[s]);   // g[s] + cost[s]
    const bool unsolved = (s != gidx);

    // histories += sel ; open -= is_unsolved*sel ; score of s -> 0 if closed
    if (lane == si) {
      hist_mask |= (1u << sj);
      if (unsolved) { open_mask &= ~(1u << sj); s_v[s] = 0.0f; }
    }

    // relax the 8 in-bounds, non-obstacle neighbors of s
    if (lane >= si - 1 && lane <= si + 1) {
      #pragma unroll
      for (int d = -1; d <= 1; ++d) {
        const int c = sj + d;
        if (c < 0 || c > 31) continue;
        if (lane == si && c == sj) continue;          // zero-center kernel
        if (!((obs_mask >> c) & 1u)) continue;        // obstacle
        const int n = (lane << 5) | c;
        const bool op = (open_mask >> c) & 1u;
        const bool take = op ? (g2v < s_g[n])         // open relax
                             : !((hist_mask >> c) & 1u); // new cell
        if (take) {
          s_g[n] = g2v;
          open_mask |= (1u << c);
          s_par[n] = s;
          const float f = __fadd_rn(__fmul_rn(0.5f, g2v),
                                    __fmul_rn(0.5f, s_ht[n]));
          s_v[n] = expf(__fdiv_rn(-f, SIZE_NORM));
        }
      }
    }
    // single wave: lockstep + in-order DS ops -> no barrier needed
  }
  __syncthreads();

  // ---- outputs: histories, then backtracked path -------------------------
  for (int j = 0; j < GRIDW; ++j) {
    const int n = (lane << 5) | j;
    out[b * HW + n] = ((hist_mask >> j) & 1u) ? 1.0f : 0.0f;
    s_v[n] = (n == gidx) ? 1.0f : 0.0f;     // reuse s_v as path marks
  }
  __syncthreads();
  if (lane == 0) {
    int loc = s_par[gidx];                  // parent of goal
    for (int t = 0; t < T_STEPS; ++t) {
      s_v[loc] = 1.0f;                      // path.at[loc].max(1)
      loc = s_par[loc];
    }
  }
  __syncthreads();
  for (int j = 0; j < GRIDW; ++j) {
    const int n = (lane << 5) | j;
    out[B * HW + b * HW + n] = s_v[n];
  }
}

// ------------------------------- launcher ----------------------------------
extern "C" void kernel_launch(void* const* d_in, const int* in_sizes, int n_in,
                              void* d_out, int out_size, void* d_ws, size_t ws_size,
                              hipStream_t stream) {
  (void)n_in; (void)out_size;
  const float* cost_maps      = (const float*)d_in[0];
  const float* start_maps     = (const float*)d_in[1];
  const float* goal_maps      = (const float*)d_in[2];
  const float* obstacles_maps = (const float*)d_in[3];
  float* out = (float*)d_out;
  const int B = in_sizes[0] / HW;           // 32

  // Reference's einsum('kij,bij->bk', loc, goal_maps) on the WMMA pipe.
  if (ws_size >= 64 * sizeof(float)) {
    goal_einsum_wmma<<<1, 32, 0, stream>>>(goal_maps, (float*)d_ws);
  }
  // Sequential A* forward: one wave32 per batch element.
  astar_wave_kernel<<<dim3(B), dim3(32), 0, stream>>>(
      cost_maps, start_maps, goal_maps, obstacles_maps, out, B);
}
